// BinaryConnect_84421877170698
// MI455X (gfx1250) — matrix operand order repaired and checked by reading, not compiled, not run on hardware
//
#include <hip/hip_runtime.h>

// ---------------------------------------------------------------------------
// BinaryConnect: out[M,N] = x[M,K] @ sign(W[K,N]) + bias[N]
// M=8192, K=4096, N=4096, fp32 in/out.
// bf16 WMMA with hi/lo split of x (sign(W) is exact in bf16).
// ---------------------------------------------------------------------------

typedef __attribute__((ext_vector_type(16))) __bf16 v16bf;
typedef __attribute__((ext_vector_type(8)))  float  v8f;

#define MDIM 8192
#define KDIM 4096
#define NDIM 4096
#define BM   128
#define BN   128
#define BK   32
#define LDA  40                  // padded LDS row pitch in bf16 elems (80B = 20 dwords)
#define NKITER (KDIM / BK)       // 128

__device__ __forceinline__ unsigned int bf16_rn_bits(float f) {
    unsigned int u = __float_as_uint(f);
    u = u + 0x7FFFu + ((u >> 16) & 1u);   // round-to-nearest-even to bf16
    return u >> 16;
}

__device__ __forceinline__ void split_hilo(float f, unsigned int& hi, unsigned int& lo) {
    unsigned int u = __float_as_uint(f);
    hi = u >> 16;                                        // truncated bf16 (exact residual)
    float r = f - __uint_as_float(u & 0xFFFF0000u);
    lo = bf16_rn_bits(r);
}

__device__ __forceinline__ unsigned int signbf(float w) {
    // w >= 0 -> +1.0 (0x3F80) ; w < 0 -> -1.0 (0xBF80)
    return 0x3F80u | ((__float_as_uint(w) >> 16) & 0x8000u);
}

union AF { uint4 q[2]; v16bf v; };

// ---- global -> register staging ------------------------------------------

__device__ __forceinline__ void load_x_tile(const float* __restrict__ x, int blockM, int k0,
                                            int tid, float4 xr[4]) {
    const int row = tid >> 3;            // 0..31 (+32*g)
    const int col = (tid & 7) << 2;      // 0,4,..,28
    const float* p = x + (size_t)(blockM + row) * KDIM + k0 + col;
#pragma unroll
    for (int g = 0; g < 4; ++g)
        xr[g] = *(const float4*)(p + (size_t)g * 32 * KDIM);
}

__device__ __forceinline__ void load_w_tile(const float* __restrict__ w, int blockN, int k0,
                                            int tid, unsigned int wp[8]) {
    const int n  = tid & 127;            // column within tile
    const int kg = (tid >> 7) << 4;      // 0 or 16
    const float* p = w + (size_t)(k0 + kg) * NDIM + blockN + n;
#pragma unroll
    for (int j = 0; j < 8; ++j) {
        float f0 = p[(size_t)(2 * j)     * NDIM];
        float f1 = p[(size_t)(2 * j + 1) * NDIM];
        wp[j] = signbf(f0) | (signbf(f1) << 16);   // packed k-pair, bf16 +-1
    }
}

// ---- register -> LDS staging ---------------------------------------------

__device__ __forceinline__ void store_x_tile(unsigned short* sAhi_, unsigned short* sAlo_,
                                             int tid, const float4 xr[4]) {
    const int row = tid >> 3;
    const int col = (tid & 7) << 2;
#pragma unroll
    for (int g = 0; g < 4; ++g) {
        unsigned int hx, lx, hy, ly, hz, lz, hw, lw;
        split_hilo(xr[g].x, hx, lx);
        split_hilo(xr[g].y, hy, ly);
        split_hilo(xr[g].z, hz, lz);
        split_hilo(xr[g].w, hw, lw);
        const int off = (row + g * 32) * LDA + col;   // byte offset multiple of 8
        *(uint2*)(sAhi_ + off) = make_uint2(hx | (hy << 16), hz | (hw << 16));
        *(uint2*)(sAlo_ + off) = make_uint2(lx | (ly << 16), lz | (lw << 16));
    }
}

__device__ __forceinline__ void store_w_tile(unsigned short* sB_, int tid,
                                             const unsigned int wp[8]) {
    const int n  = tid & 127;
    const int kg = (tid >> 7) << 4;
    uint4* d = (uint4*)(sB_ + n * LDA + kg);          // 16B aligned
    d[0] = make_uint4(wp[0], wp[1], wp[2], wp[3]);
    d[1] = make_uint4(wp[4], wp[5], wp[6], wp[7]);
}

// ---- per-K-step compute ---------------------------------------------------

__device__ __forceinline__ void compute_tiles(const unsigned short* Ah,
                                              const unsigned short* Al,
                                              const unsigned short* Bm,
                                              int lane, int wm, int wn, v8f acc[4][2]) {
    const int half = lane >> 4;      // lane half selects K sub-block
    const int sub  = lane & 15;
    AF ah[4], al[4], bfr[2];
#pragma unroll
    for (int at = 0; at < 4; ++at) {
        const int r    = wm * 64 + at * 16 + sub;
        const int base = r * LDA + half * 8;          // bytes: r*80 + half*16 (16B aligned)
        ah[at].q[0] = *(const uint4*)(Ah + base);
        ah[at].q[1] = *(const uint4*)(Ah + base + 16);
        al[at].q[0] = *(const uint4*)(Al + base);
        al[at].q[1] = *(const uint4*)(Al + base + 16);
    }
#pragma unroll
    for (int bt = 0; bt < 2; ++bt) {
        const int n    = wn * 32 + bt * 16 + sub;
        const int base = n * LDA + half * 8;          // bytes: n*80 + half*16 (16B aligned)
        bfr[bt].q[0] = *(const uint4*)(Bm + base);
        bfr[bt].q[1] = *(const uint4*)(Bm + base + 16);
    }
#pragma unroll
    for (int at = 0; at < 4; ++at) {
#pragma unroll
        for (int bt = 0; bt < 2; ++bt) {
            acc[at][bt] = __builtin_amdgcn_wmma_f32_16x16x32_bf16(
                false, ah[at].v, false, bfr[bt].v, (short)0, acc[at][bt], false, false);
            acc[at][bt] = __builtin_amdgcn_wmma_f32_16x16x32_bf16(
                false, al[at].v, false, bfr[bt].v, (short)0, acc[at][bt], false, false);
        }
    }
}

// ---- kernel ---------------------------------------------------------------

__global__ void __launch_bounds__(256)
BinaryConnect_wmma_kernel(const float* __restrict__ x, const float* __restrict__ w,
                          const float* __restrict__ bias, float* __restrict__ out) {
    __shared__ unsigned short sAhi[2][BM * LDA];
    __shared__ unsigned short sAlo[2][BM * LDA];
    __shared__ unsigned short sB  [2][BN * LDA];

    const int tid    = threadIdx.x;
    const int lane   = tid & 31;
    const int wv     = tid >> 5;     // 8 waves
    const int wm     = wv & 1;       // 2 waves along M (64 rows each)
    const int wn     = wv >> 1;      // 4 waves along N (32 cols each)
    const int blockM = blockIdx.y * BM;
    const int blockN = blockIdx.x * BN;

    // Accumulators initialized with bias[n] (same n across all 8 rows of a lane)
    v8f acc[4][2];
#pragma unroll
    for (int bt = 0; bt < 2; ++bt) {
        const float bv = bias[blockN + wn * 32 + bt * 16 + (lane & 15)];
        v8f c;
#pragma unroll
        for (int e = 0; e < 8; ++e) c[e] = bv;
#pragma unroll
        for (int at = 0; at < 4; ++at) acc[at][bt] = c;
    }

    float4       xr[4];
    unsigned int wp[8];

    // prologue: stage k-slab 0
    load_x_tile(x, blockM, 0, tid, xr);
    load_w_tile(w, blockN, 0, tid, wp);
    store_x_tile(sAhi[0], sAlo[0], tid, xr);
    store_w_tile(sB[0], tid, wp);
    __syncthreads();

    for (int kt = 0; kt < NKITER; ++kt) {
        const int cur = kt & 1;
        if (kt + 1 < NKITER) {                       // prefetch next slab into regs
            load_x_tile(x, blockM, (kt + 1) * BK, tid, xr);
            load_w_tile(w, blockN, (kt + 1) * BK, tid, wp);
        }
        compute_tiles(sAhi[cur], sAlo[cur], sB[cur], lane, wm, wn, acc);
        if (kt + 1 < NKITER) {                       // fill the other buffer
            store_x_tile(sAhi[cur ^ 1], sAlo[cur ^ 1], tid, xr);
            store_w_tile(sB[cur ^ 1], tid, wp);
        }
        __syncthreads();                             // single barrier per iter (ping-pong)
    }

    // epilogue: C/D 16x16 f32 layout -> global
#pragma unroll
    for (int at = 0; at < 4; ++at) {
#pragma unroll
        for (int bt = 0; bt < 2; ++bt) {
            const int gm = blockM + wm * 64 + at * 16 + (lane >> 4) * 8;
            const int gn = blockN + wn * 32 + bt * 16 + (lane & 15);
            float* o = out + (size_t)gm * NDIM + gn;
#pragma unroll
            for (int r = 0; r < 8; ++r)
                o[(size_t)r * NDIM] = acc[at][bt][r];
        }
    }
}

// ---- launch ---------------------------------------------------------------

extern "C" void kernel_launch(void* const* d_in, const int* in_sizes, int n_in,
                              void* d_out, int out_size, void* d_ws, size_t ws_size,
                              hipStream_t stream) {
    (void)in_sizes; (void)n_in; (void)out_size; (void)d_ws; (void)ws_size;
    const float* x    = (const float*)d_in[0];
    const float* w    = (const float*)d_in[1];
    const float* bias = (const float*)d_in[2];
    float*       out  = (float*)d_out;

    dim3 grid(NDIM / BN, MDIM / BM);   // (32, 64)
    dim3 block(256);
    BinaryConnect_wmma_kernel<<<grid, block, 0, stream>>>(x, w, bias, out);
}
